// _amsf_41463614276215
// MI455X (gfx1250) — matrix operand order repaired and checked by reading, not compiled, not run on hardware
//
#include <hip/hip_runtime.h>
#include <hip/hip_bf16.h>

typedef __attribute__((ext_vector_type(16))) _Float16 v16h;
typedef __attribute__((ext_vector_type(8)))  _Float16 v8h;
typedef __attribute__((ext_vector_type(4)))  _Float16 v4h;
typedef __attribute__((ext_vector_type(8)))  float    v8f;
typedef unsigned int u32;
typedef __attribute__((ext_vector_type(4))) u32 v4u;
typedef __attribute__((ext_vector_type(8))) int v8i;
typedef __attribute__((ext_vector_type(4))) int v4i;

#define S_SCALE 30.0f
#define MARGIN  0.4f
#define BATCH   1024
#define DIM     512
#define NCLS    100000
#define NTILES  (NCLS / 16)        // 6250 column tiles of 16 classes (exact)
#define WPB     2                  // waves per block (2 x 16 x 512 f32 = 64KB LDS)

// ---------------------------------------------------------------------------
// Kernel 1: row-normalize x -> f16, and zero the per-row exp-sum accumulators.
// ---------------------------------------------------------------------------
__global__ void k_normalize(const float* __restrict__ x,
                            _Float16* __restrict__ xn,
                            float* __restrict__ rowsum) {
    int tid  = blockIdx.x * blockDim.x + threadIdx.x;
    if (tid < BATCH) rowsum[tid] = 0.0f;
    int row  = tid >> 5;
    int lane = threadIdx.x & 31;
    if (row >= BATCH) return;

    const float4* src = (const float4*)(x + (size_t)row * DIM);   // 128 float4/row
    float ss = 0.0f;
#pragma unroll
    for (int i = 0; i < 4; ++i) {
        float4 v = src[lane + i * 32];
        ss += v.x * v.x + v.y * v.y + v.z * v.z + v.w * v.w;
    }
#pragma unroll
    for (int m = 16; m; m >>= 1) ss += __shfl_xor(ss, m, 32);
    float inv = 1.0f / fmaxf(sqrtf(ss), 1e-12f);

    v4h* dst = (v4h*)(xn + (size_t)row * DIM);
#pragma unroll
    for (int i = 0; i < 4; ++i) {
        float4 v = src[lane + i * 32];
        v4h h = { (_Float16)(v.x * inv), (_Float16)(v.y * inv),
                  (_Float16)(v.z * inv), (_Float16)(v.w * inv) };
        dst[lane + i * 32] = h;
    }
}

// ---------------------------------------------------------------------------
// Kernel 2: target logit per row: dot(xn[i], W[labels[i]]). One wave per row.
// ---------------------------------------------------------------------------
__global__ void k_target(const _Float16* __restrict__ xn,
                         const float* __restrict__ W,
                         const int* __restrict__ labels,
                         float* __restrict__ tgt) {
    int row  = (blockIdx.x * blockDim.x + threadIdx.x) >> 5;
    int lane = threadIdx.x & 31;
    if (row >= BATCH) return;
    long long c = labels[row];
    const _Float16* a = xn + (size_t)row * DIM;
    const float*    w = W + (size_t)c * DIM;
    float acc = 0.0f;
#pragma unroll
    for (int i = 0; i < 16; ++i) {
        int k = lane + i * 32;
        acc += (float)a[k] * w[k];
    }
#pragma unroll
    for (int m = 16; m; m >>= 1) acc += __shfl_xor(acc, m, 32);
    if (lane == 0) tgt[row] = acc;
}

// ---------------------------------------------------------------------------
// Kernel 3: WMMA GEMM + exp-sum. One wave owns one 16-class column tile.
// W slab (16 classes x 512 f32 = 32KB/wave) is DMA'd into LDS by the Tensor
// Data Mover (TENSOR_LOAD_TO_LDS, tracked by TENSORcnt), then converted once
// into 16 B register fragments; all 64 M-tiles of xn are swept against them,
// so W leaves HBM exactly once.
// ---------------------------------------------------------------------------
__global__ void __launch_bounds__(32 * WPB)
k_gemm_expsum(const _Float16* __restrict__ xn,
              const float* __restrict__ W,
              float* __restrict__ rowsum) {
    __shared__ float smem[WPB * 16 * DIM];        // 64 KB static LDS (f32 slabs)

    const int wave = threadIdx.x >> 5;
    const int lane = threadIdx.x & 31;
    const int nt   = blockIdx.x * WPB + wave;     // 3125 blocks x 2 waves = 6250, no tail

    float* wslab = smem + wave * 16 * DIM;

#if __has_builtin(__builtin_amdgcn_tensor_load_to_lds) && __has_builtin(__builtin_amdgcn_s_wait_tensorcnt)
    {
        // Tensor DMA descriptor (D#), per CDNA5 ISA 8.3/8.4:
        // 2-D tile: tile_dim0 = 512 elems/row, tile_dim1 = 16 rows, f32.
        unsigned long long ga = (unsigned long long)(const void*)(W + (size_t)nt * 16 * DIM);
        u32 lds_addr = (u32)(uintptr_t)(void*)wslab;   // LDS aperture: addr[31:0] = LDS offset
        v4u g0 = { 1u,                                  // count=1, user desc, no gather
                   lds_addr,                            // lds_addr [63:32]
                   (u32)ga,                             // global_addr [95:64]
                   ((u32)(ga >> 32) & 0x01FFFFFFu) | 0x80000000u }; // addr[56:32] | type=2
        v8i g1 = { (int)0x00020000u,      // workgroup_mask=0, data_size=2 (4B)
                   (int)(512u << 16),     // tensor_dim0 = 512   (bits 79:48)
                   (int)(16u  << 16),     // tensor_dim1 = 16    (bits 111:80)
                   (int)(512u << 16),     // tile_dim0   = 512   (bits 127:112)
                   16,                    // tile_dim1   = 16    (bits 143:128)
                   512,                   // tensor_dim0_stride = 512 (bits 207:160)
                   0, 0 };                // tensor_dim1_stride = 0 (2-D, unused)
        v4i g2 = { 0, 0, 0, 0 };          // 2-D tensor: dims 2+ unused
        v4i g3 = { 0, 0, 0, 0 };
#if __clang_major__ >= 23
        // amdgpu-toolchain (clang-23 / therock-10.0): 6-arg form, extra i32x8 zero-filled
        v8i gz = { 0, 0, 0, 0, 0, 0, 0, 0 };
        __builtin_amdgcn_tensor_load_to_lds(g0, g1, g2, g3, gz, 0);
#else
        // ROCm 7.2 (clang-22): 5-arg form
        __builtin_amdgcn_tensor_load_to_lds(g0, g1, g2, g3, 0);
#endif
        __builtin_amdgcn_s_wait_tensorcnt((short)0);
    }
#else
    {   // Fallback: manual staging (16*512 floats = 2048 float4 per wave)
        const float4* wsrc = (const float4*)(W + (size_t)nt * 16 * DIM);
        for (int idx = lane; idx < 2048; idx += 32)
            ((float4*)wslab)[idx] = wsrc[idx];
    }
#endif
    __syncthreads();   // order TDM/LDS writes before DS reads

    const int  l  = lane & 15;
    const bool hi = lane >= 16;

    // Build 16 B fragments (full K=512) from LDS once; lane n holds class n,
    // two 8-k chunks at k0(+8 if hi), +16. ds_load_b128 x4 + v_cvt per fragment.
    v16h Bfrag[16];
    {
        const float* bsrc = wslab + l * DIM + (hi ? 8 : 0);
#pragma unroll
        for (int kc = 0; kc < 16; ++kc) {
            v16h b;
#pragma unroll
            for (int j = 0; j < 16; ++j) b[j] = (_Float16)bsrc[kc * 32 + (j < 8 ? j : j + 8)];
            Bfrag[kc] = b;
        }
    }

    // A fragment base: lane n holds row m0+n, two 8-half chunks at k0(+8 if hi), +16
    const _Float16* abase = xn + (size_t)l * DIM + (hi ? 8 : 0);

    for (int mt0 = 0; mt0 < 64; mt0 += 4) {       // 4 M-tiles share each B frag
        v8f acc[4] = {v8f{}, v8f{}, v8f{}, v8f{}};
        const _Float16* a0 = abase + (size_t)(mt0 * 16) * DIM;
#pragma unroll
        for (int kc = 0; kc < 16; ++kc) {
            const int k0 = kc * 32;
#pragma unroll
            for (int u = 0; u < 4; ++u) {
                const _Float16* ap = a0 + (size_t)(u * 16) * DIM + k0;
                v8h alo = *(const v8h*)(ap);
                v8h ahi = *(const v8h*)(ap + 16);
                v16h A = __builtin_shufflevector(alo, ahi,
                            0,1,2,3,4,5,6,7,8,9,10,11,12,13,14,15);
                acc[u] = __builtin_amdgcn_wmma_f32_16x16x32_f16(
                            false, A, false, Bfrag[kc], (short)0, acc[u], false, false);
            }
        }
        // D layout: VGPR i holds row (i | hi*8), 16 N-values across a half-wave.
#pragma unroll
        for (int u = 0; u < 4; ++u) {
#pragma unroll
            for (int i = 0; i < 8; ++i) {
                float e = __expf(S_SCALE * acc[u][i]);
                e += __shfl_xor(e, 1, 32);
                e += __shfl_xor(e, 2, 32);
                e += __shfl_xor(e, 4, 32);
                e += __shfl_xor(e, 8, 32);
                if (l == 0) {  // lane 0 (rows 0..7) and lane 16 (rows 8..15)
                    int r = (mt0 + u) * 16 + (hi ? 8 : 0) + i;
                    atomicAdd(&rowsum[r], e);
                }
            }
        }
    }
}

// ---------------------------------------------------------------------------
// Kernel 4: loss = -mean( numer - log(exp(numer) + (sum - exp(S*t))) )
// ---------------------------------------------------------------------------
__global__ void k_finalize(const float* __restrict__ rowsum,
                           const float* __restrict__ tgt,
                           float* __restrict__ out) {
    __shared__ float red[256];
    int tid = threadIdx.x;
    float s = 0.0f;
    for (int r = tid; r < BATCH; r += 256) {
        float t     = tgt[r];
        float numer = S_SCALE * (t - MARGIN);
        float excl  = fmaxf(rowsum[r] - __expf(S_SCALE * t), 0.0f);
        float denom = __expf(numer) + excl;
        s += numer - __logf(denom);
    }
    red[tid] = s;
    __syncthreads();
    for (int off = 128; off; off >>= 1) {
        if (tid < off) red[tid] += red[tid + off];
        __syncthreads();
    }
    if (tid == 0) out[0] = -red[0] / (float)BATCH;
}

// ---------------------------------------------------------------------------
extern "C" void kernel_launch(void* const* d_in, const int* in_sizes, int n_in,
                              void* d_out, int out_size, void* d_ws, size_t ws_size,
                              hipStream_t stream) {
    const float* x      = (const float*)d_in[0];
    const int*   labels = (const int*)d_in[1];
    const float* W      = (const float*)d_in[2];

    // workspace: xn f16 (1 MB) | rowsum f32 (4 KB) | tgt f32 (4 KB)
    char*     ws     = (char*)d_ws;
    _Float16* xn     = (_Float16*)ws;
    float*    rowsum = (float*)(ws + (size_t)BATCH * DIM * sizeof(_Float16));
    float*    tgt    = rowsum + BATCH;

    k_normalize<<<(BATCH * 32 + 255) / 256, 256, 0, stream>>>(x, xn, rowsum);
    k_target  <<<(BATCH * 32 + 255) / 256, 256, 0, stream>>>(xn, W, labels, tgt);
    k_gemm_expsum<<<NTILES / WPB, 32 * WPB, 0, stream>>>(xn, W, rowsum);
    k_finalize<<<1, 256, 0, stream>>>(rowsum, tgt, (float*)d_out);
}
